// ResidualBlock_77635828843228
// MI455X (gfx1250) — compile-verified, matrix operand order checked by reading, not run on hardware
//
#include <hip/hip_runtime.h>

// ---------------------------------------------------------------------------
// CDNA5 (gfx1250, wave32) implementation of the 2-layer hetero-GraphConv.
// GEMM core: v_wmma_f32_16x16x32_bf16 (f32 accum, bf16 inputs).
// Weights pre-permuted once to bf16 WMMA layout; staged to LDS per block via
// global_load_async_to_lds_b128 (ASYNCcnt) + s_wait_asynccnt.
// ---------------------------------------------------------------------------

typedef __attribute__((ext_vector_type(16))) __bf16 v16bf;
typedef __attribute__((ext_vector_type(8)))  float  v8f;

#define DD 128
#define BN_EPS 1e-5f

// ------------------------------ zero fill ----------------------------------
__global__ void zero_kernel(float* __restrict__ p, long long n) {
  long long i = (long long)blockIdx.x * blockDim.x + threadIdx.x;
  if (i < n) p[i] = 0.0f;
}

// --------------------- edge scatter: agg[dst] += x[src] --------------------
__global__ void scatter_kernel(const float* __restrict__ xsrc,
                               const int* __restrict__ ei,
                               float* __restrict__ agg, int E) {
  long long idx = (long long)blockIdx.x * blockDim.x + threadIdx.x;
  int e = (int)(idx >> 7);
  int c = (int)(idx & 127);
  if (e < E) {
    int s = ei[e];
    int d = ei[E + e];
    unsafeAtomicAdd(&agg[(long long)d * DD + c], xsrc[(long long)s * DD + c]);
  }
}

// ---- pre-permute all 18 fused weight matrices into bf16 WMMA B layout -----
// Layout per matrix m = L*9+i:  wbf[m*32768 + ((ks*8+nt)*32 + lane)*16 + j]
__global__ void pack_weights_kernel(const float* __restrict__ Wrel,
                                    const float* __restrict__ Wroot,
                                    unsigned short* __restrict__ wbf) {
  int idx = blockIdx.x * 256 + threadIdx.x;
  if (idx >= 18 * 32768) return;
  int m = idx >> 15;
  int q = idx & 32767;
  int ks   = q >> 12;
  int nt   = (q >> 9) & 7;
  int lane = (q >> 4) & 31;
  int j    = q & 15;
  // 16-bit B operand layout (mirror of documented A layout, N indexed by lane)
  int kk = ((j < 8) ? j : j + 8) + ((lane >= 16) ? 8 : 0);
  int k  = ks * 32 + kk;                  // fused K 0..255 ([Wr ; Wo])
  int c  = nt * 16 + (lane & 15);
  const float* W = (k < 128) ? (Wrel + (size_t)m * 16384)
                             : (Wroot + (size_t)m * 16384);
  float w = W[(k & 127) * DD + c];
  __bf16 b = (__bf16)w;                   // hardware RNE convert
  wbf[idx] = __builtin_bit_cast(unsigned short, b);
}

// -------- fused GEMM h = [agg|xdst] @ [Wr;Wo]  + BN statistics -------------
// block = 128 threads = 4 waves; each wave computes a 16x128 output tile.
__global__ void __launch_bounds__(128)
gemm_bn_kernel(const float* __restrict__ agg, const float* __restrict__ xdst,
               const unsigned short* __restrict__ wmat,   // 32768 bf16, permuted
               float* __restrict__ h, float* __restrict__ stats, int Nd) {
  __shared__ __align__(32) unsigned short Bs[8 * 8 * 32 * 16]; // 64 KB
  __shared__ float lsum[128];
  __shared__ float lsq[128];

  const int tid = threadIdx.x;
  lsum[tid] = 0.0f;
  lsq[tid]  = 0.0f;

  // --- async contiguous global -> LDS copy of the permuted weight block ---
  {
    unsigned ldsbase = (unsigned)(uintptr_t)(&Bs[0]);
    // 65536 bytes total; 128 threads x 16B = 2048B per pass -> 32 passes
    for (int it = 0; it < 32; it++) {
      unsigned byteoff = ((unsigned)(it * 128 + tid)) * 16u;
      unsigned lds_addr = ldsbase + byteoff;
      unsigned long long gaddr =
          (unsigned long long)(uintptr_t)((const char*)wmat + byteoff);
      asm volatile("global_load_async_to_lds_b128 %0, %1, off"
                   :: "v"(lds_addr), "v"(gaddr) : "memory");
    }
    asm volatile("s_wait_asynccnt 0x0" ::: "memory");
  }
  __syncthreads();

  const int wave = tid >> 5;
  const int lane = tid & 31;
  const int mtile = blockIdx.x * 4 + wave;

  if (mtile * 16 < Nd) {                      // wave-uniform: EXEC stays full
    int row = mtile * 16 + (lane & 15);
    if (row >= Nd) row = Nd - 1;              // defensive clamp
    const int base0 = (lane < 16) ? 0 : 8;

    v8f acc[8] = {};                          // 16x128 f32 accumulators

    const float* rowA = agg  + (long long)row * DD;
    const float* rowX = xdst + (long long)row * DD;

#pragma unroll
    for (int ks = 0; ks < 8; ks++) {
      const float* src = (ks < 4) ? (rowA + 32 * ks) : (rowX + 32 * (ks - 4));
      // A 16-bit layout: lanes 0-15 -> K {0..7,16..23}; lanes 16-31 -> {8..15,24..31}
      float4 f0 = *reinterpret_cast<const float4*>(src + base0);
      float4 f1 = *reinterpret_cast<const float4*>(src + base0 + 4);
      float4 f2 = *reinterpret_cast<const float4*>(src + base0 + 16);
      float4 f3 = *reinterpret_cast<const float4*>(src + base0 + 20);
      v16bf av;
      av[0]  = (__bf16)f0.x; av[1]  = (__bf16)f0.y; av[2]  = (__bf16)f0.z; av[3]  = (__bf16)f0.w;
      av[4]  = (__bf16)f1.x; av[5]  = (__bf16)f1.y; av[6]  = (__bf16)f1.z; av[7]  = (__bf16)f1.w;
      av[8]  = (__bf16)f2.x; av[9]  = (__bf16)f2.y; av[10] = (__bf16)f2.z; av[11] = (__bf16)f2.w;
      av[12] = (__bf16)f3.x; av[13] = (__bf16)f3.y; av[14] = (__bf16)f3.z; av[15] = (__bf16)f3.w;

#pragma unroll
      for (int nt = 0; nt < 8; nt++) {
        v16bf b = *reinterpret_cast<v16bf*>(&Bs[((ks * 8 + nt) * 32 + lane) * 16]);
        acc[nt] = __builtin_amdgcn_wmma_f32_16x16x32_bf16(
            /*neg_a=*/false, av, /*neg_b=*/false, b,
            /*c_mod=*/(short)0, acc[nt], /*reuse_a=*/false, /*reuse_b=*/false);
      }
    }

    // Store h and accumulate per-channel BN partials.
    const int half = lane >> 4;   // C/D layout: VGPR v -> M = v + 8*half
    const int ln   = lane & 15;
#pragma unroll
    for (int nt = 0; nt < 8; nt++) {
      float s = 0.0f, qs = 0.0f;
#pragma unroll
      for (int v = 0; v < 8; v++) {
        float x = acc[nt][v];
        int r = mtile * 16 + v + 8 * half;
        if (r < Nd) h[(long long)r * DD + nt * 16 + ln] = x;
        s  += x;
        qs += x * x;
      }
      atomicAdd(&lsum[nt * 16 + ln], s);    // ds_add_f32
      atomicAdd(&lsq[nt * 16 + ln], qs);
    }
  }
  __syncthreads();

  // one global f32 atomic per channel per block
  unsafeAtomicAdd(&stats[tid],       lsum[tid]);
  unsafeAtomicAdd(&stats[128 + tid], lsq[tid]);
}

// ---------------- BN apply + accumulate into destination output ------------
__global__ void bn_apply_kernel(const float* __restrict__ h,
                                const float* __restrict__ stats,
                                const float* __restrict__ gamma,
                                const float* __restrict__ beta,
                                float* __restrict__ out, int Nd) {
  long long idx = (long long)blockIdx.x * blockDim.x + threadIdx.x;
  long long total = (long long)Nd * DD;
  if (idx < total) {
    int c = (int)(idx & 127);
    float invN = 1.0f / (float)Nd;
    float mu  = stats[c] * invN;
    float var = stats[128 + c] * invN - mu * mu;
    float inv = rsqrtf(var + BN_EPS);
    out[idx] += gamma[c] * (h[idx] - mu) * inv + beta[c];
  }
}

// ------------------------------- residual ----------------------------------
__global__ void add_kernel(const float* __restrict__ a, const float* __restrict__ b,
                           float* __restrict__ o, long long n) {
  long long i = (long long)blockIdx.x * blockDim.x + threadIdx.x;
  if (i < n) o[i] = a[i] + b[i];
}

// ===========================================================================
extern "C" void kernel_launch(void* const* d_in, const int* in_sizes, int n_in,
                              void* d_out, int out_size, void* d_ws, size_t ws_size,
                              hipStream_t stream) {
  (void)n_in; (void)out_size; (void)ws_size;

  const float* x_in[3] = {(const float*)d_in[0], (const float*)d_in[1], (const float*)d_in[2]};
  const int* ei[9];
  int E[9];
  for (int i = 0; i < 9; i++) {
    ei[i] = (const int*)d_in[3 + i];
    E[i]  = in_sizes[3 + i] / 2;
  }
  const float* W_rel  = (const float*)d_in[12];
  // d_in[13] = b_rel: cancels exactly inside BatchNorm -> unused.
  const float* W_root = (const float*)d_in[14];
  const float* gamma  = (const float*)d_in[15];
  const float* beta   = (const float*)d_in[16];

  long long Nn[3] = { in_sizes[0] / DD, in_sizes[1] / DD, in_sizes[2] / DD };
  static const int SRC[9] = {0, 1, 0, 2, 1, 2, 0, 1, 2};
  static const int DST[9] = {1, 0, 2, 0, 2, 1, 0, 1, 2};

  // ---- workspace carve-up (floats) ----
  float* ws = (float*)d_ws;
  size_t off = 0;
  float* cur[3]; float* nxt[3];
  for (int t = 0; t < 3; t++) { cur[t] = ws + off; off += (size_t)Nn[t] * DD; }
  for (int t = 0; t < 3; t++) { nxt[t] = ws + off; off += (size_t)Nn[t] * DD; }
  long long Nmax = Nn[0] > Nn[1] ? Nn[0] : Nn[1];
  if (Nn[2] > Nmax) Nmax = Nn[2];
  float* aggbuf = ws + off; off += (size_t)Nmax * DD;
  float* hbuf   = ws + off; off += (size_t)Nmax * DD;
  float* stats  = ws + off; off += 256;
  unsigned short* wbf = (unsigned short*)(ws + off);
  off += (size_t)(18 * 32768) / 2;          // 18 matrices x 32768 bf16

  auto zero = [&](float* p, long long n) {
    zero_kernel<<<dim3((unsigned)((n + 255) / 256)), dim3(256), 0, stream>>>(p, n);
  };

  // Pre-permute weights (once; ~1.2 MB, stays L2-resident)
  pack_weights_kernel<<<dim3((18 * 32768 + 255) / 256), dim3(256), 0, stream>>>(
      W_rel, W_root, wbf);

  for (int L = 0; L < 2; L++) {
    const float* xin[3];
    float* xout[3];
    for (int t = 0; t < 3; t++) {
      xin[t]  = (L == 0) ? x_in[t] : (const float*)cur[t];
      xout[t] = (L == 0) ? cur[t] : nxt[t];
    }
    for (int t = 0; t < 3; t++) zero(xout[t], Nn[t] * DD);

    for (int i = 0; i < 9; i++) {
      int s = SRC[i], d = DST[i];
      long long Nd = Nn[d];

      zero(aggbuf, Nd * DD);
      long long sthreads = (long long)E[i] * DD;
      scatter_kernel<<<dim3((unsigned)((sthreads + 255) / 256)), dim3(256), 0, stream>>>(
          xin[s], ei[i], aggbuf, E[i]);

      zero(stats, 256);

      const unsigned short* wmat = wbf + (size_t)(L * 9 + i) * 32768;
      int mtiles = (int)((Nd + 15) / 16);
      gemm_bn_kernel<<<dim3((unsigned)((mtiles + 3) / 4)), dim3(128), 0, stream>>>(
          aggbuf, xin[d], wmat, hbuf, stats, (int)Nd);

      long long at = Nd * DD;
      bn_apply_kernel<<<dim3((unsigned)((at + 255) / 256)), dim3(256), 0, stream>>>(
          hbuf, stats, gamma + (size_t)(L * 9 + i) * DD, beta + (size_t)(L * 9 + i) * DD,
          xout[d], (int)Nd);
    }
  }

  // residual: d_out = layer2 output + original inputs (atom ‖ bond ‖ global)
  float* out = (float*)d_out;
  size_t ofs = 0;
  for (int t = 0; t < 3; t++) {
    long long n = Nn[t] * DD;
    add_kernel<<<dim3((unsigned)((n + 255) / 256)), dim3(256), 0, stream>>>(
        nxt[t], x_in[t], out + ofs, n);
    ofs += (size_t)n;
  }
}
